// SpGraphAttentionLayer_79491254714922
// MI455X (gfx1250) — compile-verified, matrix operand order checked by reading, not run on hardware
//
#include <hip/hip_runtime.h>
#include <hip/hip_bf16.h>

#define F 128           // feature dim (F_IN == F_OUT == 128)
#define ALPHA 0.2f
#define LN_EPS 1e-5f

typedef __attribute__((ext_vector_type(16))) __bf16 v16bf;
typedef __attribute__((ext_vector_type(8)))  __bf16 v8bf;
typedef __attribute__((ext_vector_type(8)))  float  v8f;

// ---- monotonic float<->uint key for atomicMax over signed floats ----
__device__ __forceinline__ unsigned fkey(float f) {
    unsigned u = __float_as_uint(f);
    return (u & 0x80000000u) ? ~u : (u | 0x80000000u);
}
__device__ __forceinline__ float fdecode(unsigned k) {
    return (k & 0x80000000u) ? __uint_as_float(k & 0x7fffffffu)
                             : __uint_as_float(~k);
}

// ---------------- K0: init accumulators ----------------
__global__ void k_init(float* __restrict__ out, float* __restrict__ rowsum,
                       unsigned* __restrict__ gkey, int N) {
    long long i = (long long)blockIdx.x * blockDim.x + threadIdx.x;
    long long total = (long long)N * F;
    long long stride = (long long)gridDim.x * blockDim.x;
    for (long long idx = i; idx < total; idx += stride) {
        out[idx] = 0.0f;
        if (idx < N) rowsum[idx] = 1e-8f;
    }
    if (i == 0) *gkey = 0u;
}

// ---------------- K1: x -> bf16 ----------------
__global__ void k_cvt_x(const float* __restrict__ x, __bf16* __restrict__ xb, int N) {
    long long i = (long long)blockIdx.x * blockDim.x + threadIdx.x;
    long long total = (long long)N * F;
    long long stride = (long long)gridDim.x * blockDim.x;
    for (long long idx = i; idx < total; idx += stride)
        xb[idx] = (__bf16)x[idx];
}

// ---------------- K2: W -> W^T bf16 (so B-columns are contiguous) ----------------
__global__ void k_prep_w(const float* __restrict__ W, __bf16* __restrict__ Wt) {
    int i = blockIdx.x * blockDim.x + threadIdx.x;
    if (i < F * F) {
        int k = i / F, f = i % F;
        Wt[f * F + k] = (__bf16)W[k * F + f];
    }
}

// ---------------- K3: WMMA GEMM h = x@W + bias, fused s_src/s_dst = h@a ----------------
// block = 256 threads = 8 waves; wave w computes rows [blk*128 + w*16, +16) x all 128 cols.
__global__ void __launch_bounds__(256)
k_gemm(const __bf16* __restrict__ xb, const __bf16* __restrict__ Wt,
       const float* __restrict__ bias, const float* __restrict__ avec,
       float* __restrict__ h, float* __restrict__ ssrc, float* __restrict__ sdst,
       int N) {
    __shared__ __bf16 sW[F * F];   // 32 KB: W^T, row f = column f of W, 128 bf16 each

    const int tid = threadIdx.x;
    for (int i = tid * 8; i < F * F; i += 256 * 8)
        *(v8bf*)(sW + i) = *(const v8bf*)(Wt + i);
    __syncthreads();

    const int lane  = tid & 31;
    const int wave  = tid >> 5;
    const int n     = lane & 15;   // B/C column within tile, A row within tile
    const int khalf = lane >> 4;   // which K-half this lane holds (ISA 16-bit layout)
    const int tileRow = blockIdx.x * 128 + wave * 16;
    const int rowA    = tileRow + n;           // A-matrix row for this lane

    // 8 accumulators (16x16 f32 each), preloaded with bias[col]
    v8f acc[8];
#pragma unroll
    for (int t = 0; t < 8; t++) {
        float b = bias[t * 16 + n];
#pragma unroll
        for (int r = 0; r < 8; r++) acc[t][r] = b;
    }

#pragma unroll
    for (int kk = 0; kk < 4; kk++) {
        const int k0 = kk * 32 + khalf * 8;
        v16bf af;
        if (rowA < N) {
            v8bf lo = *(const v8bf*)(xb + (long long)rowA * F + k0);
            v8bf hi = *(const v8bf*)(xb + (long long)rowA * F + k0 + 16);
#pragma unroll
            for (int i = 0; i < 8; i++) { af[i] = lo[i]; af[8 + i] = hi[i]; }
        } else {
#pragma unroll
            for (int i = 0; i < 16; i++) af[i] = (__bf16)0.0f;
        }
#pragma unroll
        for (int t = 0; t < 8; t++) {
            v8bf blo = *(const v8bf*)(sW + (t * 16 + n) * F + k0);
            v8bf bhi = *(const v8bf*)(sW + (t * 16 + n) * F + k0 + 16);
            v16bf bfr;
#pragma unroll
            for (int i = 0; i < 8; i++) { bfr[i] = blo[i]; bfr[8 + i] = bhi[i]; }
            acc[t] = __builtin_amdgcn_wmma_f32_16x16x32_bf16(
                false, af, false, bfr, (short)0, acc[t], false, false);
        }
    }

    // store h rows + fused attention projections s_src = h@a[:F], s_dst = h@a[F:]
    float ps[8], pd[8];
#pragma unroll
    for (int r = 0; r < 8; r++) { ps[r] = 0.0f; pd[r] = 0.0f; }
#pragma unroll
    for (int t = 0; t < 8; t++) {
        float as = avec[t * 16 + n];
        float ad = avec[F + t * 16 + n];
#pragma unroll
        for (int r = 0; r < 8; r++) {
            int row = tileRow + khalf * 8 + r;   // C/D layout: lanes>=16 hold rows 8..15
            float v = acc[t][r];
            if (row < N) h[(long long)row * F + t * 16 + n] = v;
            ps[r] += v * as;
            pd[r] += v * ad;
        }
    }
    // reduce over the 16 lanes sharing the same row group (masks<16 stay in-group)
#pragma unroll
    for (int r = 0; r < 8; r++) {
#pragma unroll
        for (int m = 1; m < 16; m <<= 1) {
            ps[r] += __shfl_xor(ps[r], m, 32);
            pd[r] += __shfl_xor(pd[r], m, 32);
        }
    }
    if (n == 0) {
#pragma unroll
        for (int r = 0; r < 8; r++) {
            int row = tileRow + khalf * 8 + r;
            if (row < N) { ssrc[row] = ps[r]; sdst[row] = pd[r]; }
        }
    }
}

// ---------------- K4: edge pass 1 -- leaky scores + global max ----------------
__global__ void k_edge1(const float* __restrict__ ssrc, const float* __restrict__ sdst,
                        const int* __restrict__ src, const int* __restrict__ dst,
                        float* __restrict__ ebuf, unsigned* __restrict__ gkey, int E) {
    int i = blockIdx.x * blockDim.x + threadIdx.x;
    int stride = gridDim.x * blockDim.x;
    float lmax = -3.4e38f;
    for (int e = i; e < E; e += stride) {
        float t = ssrc[src[e]] + sdst[dst[e]];
        t = (t > 0.0f) ? t : ALPHA * t;
        ebuf[e] = t;
        lmax = fmaxf(lmax, t);
    }
#pragma unroll
    for (int m = 16; m >= 1; m >>= 1) lmax = fmaxf(lmax, __shfl_xor(lmax, m, 32));
    if ((threadIdx.x & 31) == 0) atomicMax(gkey, fkey(lmax));
}

// ---------------- K5: edge pass 2 -- exp + scatter-add (one wave per edge) ----------------
__global__ void k_edge2(const float* __restrict__ ebuf, const unsigned* __restrict__ gkey,
                        const int* __restrict__ src, const int* __restrict__ dst,
                        const float* __restrict__ h, float* __restrict__ rowsum,
                        float* __restrict__ out, int E) {
    const int lane = threadIdx.x & 31;
    const int wid  = blockIdx.x * (blockDim.x >> 5) + (threadIdx.x >> 5);
    const int nw   = gridDim.x * (blockDim.x >> 5);
    const float gmax = fdecode(*gkey);
    for (int e = wid; e < E; e += nw) {
        int s = src[e], d = dst[e];
        float w = expf(ebuf[e] - gmax);
        if (lane == 0) atomicAdd(&rowsum[s], w);
        const float4 hv = *(const float4*)(h + (long long)d * F + lane * 4);
        float* o = out + (long long)s * F + lane * 4;
        atomicAdd(o + 0, w * hv.x);
        atomicAdd(o + 1, w * hv.y);
        atomicAdd(o + 2, w * hv.z);
        atomicAdd(o + 3, w * hv.w);
    }
}

// ---------------- K6: normalize + residual + LayerNorm + ELU (one wave per node) ----------------
__global__ void k_final(float* __restrict__ out, const float* __restrict__ x,
                        const float* __restrict__ rowsum,
                        const float* __restrict__ gamma, const float* __restrict__ beta,
                        int N) {
    const int lane = threadIdx.x & 31;
    const int node = blockIdx.x * (blockDim.x >> 5) + (threadIdx.x >> 5);
    if (node >= N) return;
    const long long base = (long long)node * F + lane * 4;
    float4 acc = *(const float4*)(out + base);
    float4 xi  = *(const float4*)(x + base);
    float inv = 1.0f / rowsum[node];
    float v[4];
    v[0] = acc.x * inv + xi.x;
    v[1] = acc.y * inv + xi.y;
    v[2] = acc.z * inv + xi.z;
    v[3] = acc.w * inv + xi.w;
    float s  = v[0] + v[1] + v[2] + v[3];
    float ss = v[0] * v[0] + v[1] * v[1] + v[2] * v[2] + v[3] * v[3];
#pragma unroll
    for (int m = 16; m >= 1; m >>= 1) {
        s  += __shfl_xor(s,  m, 32);
        ss += __shfl_xor(ss, m, 32);
    }
    const float mu   = s * (1.0f / F);
    const float var  = ss * (1.0f / F) - mu * mu;
    const float rstd = rsqrtf(var + LN_EPS);
    float4 res;
    float* rp = &res.x;
#pragma unroll
    for (int j = 0; j < 4; j++) {
        int f = lane * 4 + j;
        float t = (v[j] - mu) * rstd * gamma[f] + beta[f];
        rp[j] = (t > 0.0f) ? t : expm1f(t);   // ELU(alpha=1)
    }
    *(float4*)(out + base) = res;
}

extern "C" void kernel_launch(void* const* d_in, const int* in_sizes, int n_in,
                              void* d_out, int out_size, void* d_ws, size_t ws_size,
                              hipStream_t stream) {
    const float* x     = (const float*)d_in[0];
    const int*   edge  = (const int*)d_in[1];
    const float* W     = (const float*)d_in[2];
    const float* a     = (const float*)d_in[3];
    const float* bias  = (const float*)d_in[4];
    const float* gamma = (const float*)d_in[5];
    const float* beta  = (const float*)d_in[6];
    const int N = in_sizes[0] / F;
    const int E = in_sizes[1] / 2;
    const int* src = edge;
    const int* dst = edge + E;
    float* out = (float*)d_out;

    // workspace carve-out (256B aligned): ~84 MB total
    char* ws = (char*)d_ws;
    size_t off = 0;
    auto carve = [&](size_t bytes) -> void* {
        off = (off + 255) & ~(size_t)255;
        void* p = ws + off;
        off += bytes;
        return p;
    };
    __bf16*   xb     = (__bf16*)carve((size_t)N * F * sizeof(__bf16));
    __bf16*   Wt     = (__bf16*)carve((size_t)F * F * sizeof(__bf16));
    float*    h      = (float*)carve((size_t)N * F * sizeof(float));
    float*    ssrc   = (float*)carve((size_t)N * sizeof(float));
    float*    sdst   = (float*)carve((size_t)N * sizeof(float));
    float*    ebuf   = (float*)carve((size_t)E * sizeof(float));
    float*    rowsum = (float*)carve((size_t)N * sizeof(float));
    unsigned* gkey   = (unsigned*)carve(sizeof(unsigned));

    k_init  <<<2048, 256, 0, stream>>>(out, rowsum, gkey, N);
    k_cvt_x <<<2048, 256, 0, stream>>>(x, xb, N);
    k_prep_w<<<(F * F + 255) / 256, 256, 0, stream>>>(W, Wt);
    k_gemm  <<<(N + 127) / 128, 256, 0, stream>>>(xb, Wt, bias, a, h, ssrc, sdst, N);
    k_edge1 <<<1024, 256, 0, stream>>>(ssrc, sdst, src, dst, ebuf, gkey, E);
    k_edge2 <<<4096, 256, 0, stream>>>(ebuf, gkey, src, dst, h, rowsum, out, E);
    k_final <<<(N + 7) / 8, 256, 0, stream>>>(out, x, rowsum, gamma, beta, N);
}
